// DenseEdgeConv_25383256719485
// MI455X (gfx1250) — matrix operand order repaired and checked by reading, not compiled, not run on hardware
//
#include <hip/hip_runtime.h>

#define BATCH 8
#define NPTS  4096
#define DIM   128
#define KNN   16
#define OUTC  384

typedef __bf16 bf16_t;
typedef __attribute__((ext_vector_type(16))) __bf16 v16bf;
typedef __attribute__((ext_vector_type(8)))  float  v8f;
typedef int v4i_t __attribute__((ext_vector_type(4)));
typedef int v8i_t __attribute__((ext_vector_type(8)));

// plain 16-byte chunk usable inside a union (HIP's uint4 has ctors)
struct __attribute__((aligned(16))) uq4 { unsigned int a, b, c, d; };

// ---------------------------------------------------------------------------
// Kernel 0: one-time fp32 -> bf16 conversion of x (table feeds the TDM gather)
// ---------------------------------------------------------------------------
__global__ __launch_bounds__(256) void cvt_kernel(const float* __restrict__ x,
                                                  bf16_t* __restrict__ xb)
{
    const size_t i = ((size_t)blockIdx.x * 256 + threadIdx.x) * 4;
    const float4 v = *(const float4*)(x + i);
    bf16_t* d = xb + i;
    d[0] = (bf16_t)v.x; d[1] = (bf16_t)v.y;
    d[2] = (bf16_t)v.z; d[3] = (bf16_t)v.w;
}

// ---------------------------------------------------------------------------
// Kernel 1: brute-force KNN (16 nearest, excluding self), per-batch indices
// ---------------------------------------------------------------------------
__global__ __launch_bounds__(256) void knn_kernel(const float* __restrict__ pos,
                                                  int* __restrict__ knn_out)
{
    const int tid = threadIdx.x;
    const int b   = blockIdx.x >> 4;                 // 16 tiles of 256 per batch
    const int n   = ((blockIdx.x & 15) << 8) + tid;

    const float* pb = pos + (size_t)b * NPTS * 3;
    const float px = pb[n * 3 + 0];
    const float py = pb[n * 3 + 1];
    const float pz = pb[n * 3 + 2];

    float dist[KNN];
    int   nidx[KNN];
#pragma unroll
    for (int j = 0; j < KNN; ++j) { dist[j] = 3.0e38f; nidx[j] = 0; }

    __shared__ float sp[256 * 3];
    for (int m0 = 0; m0 < NPTS; m0 += 256) {
        __syncthreads();
        sp[tid * 3 + 0] = pb[(m0 + tid) * 3 + 0];
        sp[tid * 3 + 1] = pb[(m0 + tid) * 3 + 1];
        sp[tid * 3 + 2] = pb[(m0 + tid) * 3 + 2];
        __syncthreads();
        for (int mm = 0; mm < 256; ++mm) {
            const int m = m0 + mm;
            const float dx = px - sp[mm * 3 + 0];
            const float dy = py - sp[mm * 3 + 1];
            const float dz = pz - sp[mm * 3 + 2];
            const float d2 = dx * dx + dy * dy + dz * dz;
            if (m != n && d2 < dist[KNN - 1]) {
#pragma unroll
                for (int j = KNN - 1; j >= 1; --j) {
                    const bool lt_j = d2 < dist[j];
                    const bool lt_p = d2 < dist[j - 1];
                    const float nd = lt_j ? (lt_p ? dist[j - 1] : d2) : dist[j];
                    const int   ni = lt_j ? (lt_p ? nidx[j - 1] : m ) : nidx[j];
                    dist[j] = nd; nidx[j] = ni;
                }
                if (d2 < dist[0]) { dist[0] = d2; nidx[0] = m; }
            }
        }
    }
    int* ko = knn_out + ((size_t)b * NPTS + n) * KNN;
#pragma unroll
    for (int j = 0; j < KNN; ++j) ko[j] = nidx[j];
}

// ---------------------------------------------------------------------------
// Kernel 2: fused dense edge-conv: TDM gather + 4 WMMA GEMM layers + K-max
// ---------------------------------------------------------------------------
#define KP1 264   // padded K-pitch (elems) of layer-1 weights (K=256)
#define KP2 200   // layer-2 (K=192)
#define KP3 264   // layer-3 (K=256)
#define KP4 328   // layer-4 (K=320)
#define NBP 136   // neighbor-feature row pitch (128 data + 8 pad elems = 272 B)
#define HP  72    // hidden row pitch (64 + pad)
#define WAVES 8
#define ITERS 16
#define NBLK  256

__device__ __attribute__((always_inline)) inline
v16bf load_a16x32(const bf16_t* base, int pitch, int k0, int lane)
{
    const int row = lane & 15;
    const int kh  = (lane & 16) ? 8 : 0;
    const bf16_t* p = base + row * pitch + k0 + kh;
    union { uq4 q[2]; v16bf v; } u;
    u.q[0] = *(const uq4*)(p);
    u.q[1] = *(const uq4*)(p + 16);
    return u.v;
}

__device__ __attribute__((always_inline)) inline
v16bf load_b32x16(const bf16_t* wbase, int pitch, int n0, int k0, int lane)
{
    const int col = lane & 15;
    const int kh  = (lane & 16) ? 8 : 0;
    const bf16_t* p = wbase + (n0 + col) * pitch + k0 + kh;
    union { uq4 q[2]; v16bf v; } u;
    u.q[0] = *(const uq4*)(p);
    u.q[1] = *(const uq4*)(p + 16);
    return u.v;
}

__device__ __attribute__((always_inline)) inline
void gemm_acc(const bf16_t* abase, int apitch, int alen, int wrow,
              const bf16_t* wbase, int wpitch, int lane, v8f acc[4])
{
#pragma unroll
    for (int k0 = 0; k0 < alen; k0 += 32) {
        const v16bf a = load_a16x32(abase, apitch, k0, lane);
#pragma unroll
        for (int nt = 0; nt < 4; ++nt) {
            const v16bf bm = load_b32x16(wbase, wpitch, nt * 16, wrow + k0, lane);
            acc[nt] = __builtin_amdgcn_wmma_f32_16x16x32_bf16(
                false, a, false, bm, (short)0, acc[nt], false, false);
        }
    }
}

__device__ __attribute__((always_inline)) inline
void epilogue(v8f acc[4], const float* __restrict__ bias, bf16_t* hdst,
              float* __restrict__ outp, int outOff, bool relu, int lane)
{
    const int col   = lane & 15;
    const int rhalf = (lane & 16) ? 8 : 0;
#pragma unroll
    for (int nt = 0; nt < 4; ++nt) {
        const float bv = bias[nt * 16 + col];
        float m = -3.0e38f;
#pragma unroll
        for (int r = 0; r < 8; ++r) {
            float v = acc[nt][r] + bv;
            if (relu) v = fmaxf(v, 0.0f);
            if (hdst) hdst[(r + rhalf) * HP + nt * 16 + col] = (bf16_t)v;
            m = fmaxf(m, v);
        }
        m = fmaxf(m, __shfl_xor(m, 16, 32));          // combine the two M-halves
        if (lane < 16) outp[outOff + nt * 16 + lane] = m;
    }
}

__global__ __launch_bounds__(256) void edgeconv_kernel(
    const float*  __restrict__ x,
    const bf16_t* __restrict__ xb,
    const float* __restrict__ Wf,  const float* __restrict__ b0,
    const float* __restrict__ Wm1, const float* __restrict__ b1,
    const float* __restrict__ Wm2, const float* __restrict__ b2,
    const float* __restrict__ Wl,  const float* __restrict__ b3,
    const int*   __restrict__ knn_idx,
    float* __restrict__ out)
{
    __shared__ __attribute__((aligned(16))) bf16_t sW1[64 * KP1];
    __shared__ __attribute__((aligned(16))) bf16_t sW2[64 * KP2];
    __shared__ __attribute__((aligned(16))) bf16_t sW3[64 * KP3];
    __shared__ __attribute__((aligned(16))) bf16_t sW4[64 * KP4];
    __shared__ __attribute__((aligned(16))) bf16_t sX [WAVES][NBP];
    __shared__ __attribute__((aligned(16))) bf16_t sNB[WAVES][KNN * NBP];
    __shared__ __attribute__((aligned(16))) bf16_t sH [WAVES][3 * 16 * HP];

    const int tid  = threadIdx.x;
    const int lane = tid & 31;
    const int wave = tid >> 5;

    // ---- stage weights: bf16, column-major, padded pitch ----
    for (int t = tid; t < 64 * 256; t += 256) {
        const int n = t >> 8, k = t & 255;
        const float w = (k < 128) ? (Wf[k * 64 + n] - Wf[(k + 256) * 64 + n])
                                  : (Wf[k * 64 + n] + Wf[(k + 128) * 64 + n]);
        sW1[n * KP1 + k] = (bf16_t)w;
    }
    for (int t = tid; t < 64 * 192; t += 256) { const int n = t / 192, k = t - n * 192; sW2[n * KP2 + k] = (bf16_t)Wm1[k * 64 + n]; }
    for (int t = tid; t < 64 * 256; t += 256) { const int n = t >> 8,  k = t & 255;     sW3[n * KP3 + k] = (bf16_t)Wm2[k * 64 + n]; }
    for (int t = tid; t < 64 * 320; t += 256) { const int n = t / 320, k = t - n * 320; sW4[n * KP4 + k] = (bf16_t)Wl [k * 64 + n]; }
    __syncthreads();

    bf16_t* sXw  = sX[wave];
    bf16_t* sNBw = sNB[wave];
    bf16_t* sH1  = sH[wave];
    bf16_t* sH2  = sH1 + 16 * HP;
    bf16_t* sH3  = sH2 + 16 * HP;

    for (int it = 0; it < ITERS; ++it) {
        const int p = (blockIdx.x * ITERS + it) * WAVES + wave;  // point id
        const int b = p >> 12;                                   // N = 4096

        // ---- TDM gather: 16 neighbor bf16 rows -> LDS (padded pitch) ----
        {
            const int4* ip4 = (const int4*)(knn_idx + (size_t)p * KNN);
            const int4 i0 = ip4[0], i1 = ip4[1], i2 = ip4[2], i3 = ip4[3];

            const unsigned ldsa = (unsigned)(size_t)sNBw;
            const unsigned long long ga =
                (unsigned long long)(size_t)(xb + (size_t)b * NPTS * DIM);

            v4i_t g0;  // count=1 | gather_mode(16-bit idx) | lds_addr | global_addr | type=2
            g0[0] = __builtin_amdgcn_readfirstlane((int)(1u | (1u << 31)));
            g0[1] = __builtin_amdgcn_readfirstlane((int)ldsa);
            g0[2] = __builtin_amdgcn_readfirstlane((int)(ga & 0xFFFFFFFFull));
            g0[3] = __builtin_amdgcn_readfirstlane((int)((unsigned)(ga >> 32) | 0x80000000u));

            v8i_t g1;  // data_size=8B, pad 4 dw every 64 dw, dims/strides in 8B units
            g1[0] = (3 << 16) | (1 << 20) | (5 << 22) | (3 << 25);
            g1[1] = 32 << 16;          // tensor_dim0 = 32   (bits 79:48, low half)
            g1[2] = 4096 << 16;        // tensor_dim1 = 4096 (bits 111:80, low half)
            g1[3] = 32 << 16;          // tile_dim0 = 32     (bits 127:112)
            g1[4] = 16;                // tile_dim1 = #valid row indices = 16
            g1[5] = 32;                // tensor_dim0_stride = 32 (row pitch, 8B units)
            g1[6] = 0;
            g1[7] = 0;

            v4i_t g2, g3;              // 16 x 16-bit row indices
            g2[0] = __builtin_amdgcn_readfirstlane((i0.y << 16) | (i0.x & 0xFFFF));
            g2[1] = __builtin_amdgcn_readfirstlane((i0.w << 16) | (i0.z & 0xFFFF));
            g2[2] = __builtin_amdgcn_readfirstlane((i1.y << 16) | (i1.x & 0xFFFF));
            g2[3] = __builtin_amdgcn_readfirstlane((i1.w << 16) | (i1.z & 0xFFFF));
            g3[0] = __builtin_amdgcn_readfirstlane((i2.y << 16) | (i2.x & 0xFFFF));
            g3[1] = __builtin_amdgcn_readfirstlane((i2.w << 16) | (i2.z & 0xFFFF));
            g3[2] = __builtin_amdgcn_readfirstlane((i3.y << 16) | (i3.x & 0xFFFF));
            g3[3] = __builtin_amdgcn_readfirstlane((i3.w << 16) | (i3.z & 0xFFFF));

            asm volatile("tensor_load_to_lds %0, %1, %2, %3"
                         :: "s"(g0), "s"(g1), "s"(g2), "s"(g3)
                         : "memory");
        }

        // stage center feature (bf16 once; A reads it with pitch 0); keep fp32 copy
        const float4* xp4 = (const float4*)(x + (size_t)p * DIM);
        const float4 xv = xp4[lane];
        {
            bf16_t* d = sXw + lane * 4;
            d[0] = (bf16_t)xv.x; d[1] = (bf16_t)xv.y;
            d[2] = (bf16_t)xv.z; d[3] = (bf16_t)xv.w;
        }

        float* outp = out + (size_t)p * OUTC;

        { // layer 1: h1 = relu([x | nb] @ W1c + b0)       (K = 256)
            v8f acc[4] = {};
            gemm_acc(sXw,  0,   128, 0,   sW1, KP1, lane, acc);   // overlaps the DMA
            asm volatile("s_wait_tensorcnt 0x0" ::: "memory");    // NB rows ready
            gemm_acc(sNBw, NBP, 128, 128, sW1, KP1, lane, acc);
            epilogue(acc, b0, sH1, outp, 192, true, lane);
        }
        { // layer 2: h2 = relu([h1 | x] @ Wm1 + b1)       (K = 192)
            v8f acc[4] = {};
            gemm_acc(sH1, HP, 64,  0,  sW2, KP2, lane, acc);
            gemm_acc(sXw, 0,  128, 64, sW2, KP2, lane, acc);
            epilogue(acc, b1, sH2, outp, 128, true, lane);
        }
        { // layer 3: h3 = relu([h2 | h1 | x] @ Wm2 + b2)  (K = 256)
            v8f acc[4] = {};
            gemm_acc(sH2, HP, 64,  0,   sW3, KP3, lane, acc);
            gemm_acc(sH1, HP, 64,  64,  sW3, KP3, lane, acc);
            gemm_acc(sXw, 0,  128, 128, sW3, KP3, lane, acc);
            epilogue(acc, b2, sH3, outp, 64, true, lane);
        }
        { // layer 4: h4 = [h3 | h2 | h1 | x] @ Wl + b3    (K = 320, no relu)
            v8f acc[4] = {};
            gemm_acc(sH3, HP, 64,  0,   sW4, KP4, lane, acc);
            gemm_acc(sH2, HP, 64,  64,  sW4, KP4, lane, acc);
            gemm_acc(sH1, HP, 64,  128, sW4, KP4, lane, acc);
            gemm_acc(sXw, 0,  128, 192, sW4, KP4, lane, acc);
            epilogue(acc, b3, nullptr, outp, 0, false, lane);
        }

        // x passthrough (max over K of x_tiled == x), channels 256..383
        outp[256 + lane * 4 + 0] = xv.x;
        outp[256 + lane * 4 + 1] = xv.y;
        outp[256 + lane * 4 + 2] = xv.z;
        outp[256 + lane * 4 + 3] = xv.w;
    }
}

// ---------------------------------------------------------------------------
extern "C" void kernel_launch(void* const* d_in, const int* in_sizes, int n_in,
                              void* d_out, int out_size, void* d_ws, size_t ws_size,
                              hipStream_t stream)
{
    const float* x   = (const float*)d_in[0];
    const float* pos = (const float*)d_in[1];
    const float* Wf  = (const float*)d_in[2];
    const float* b0  = (const float*)d_in[3];
    const float* Wm1 = (const float*)d_in[4];
    const float* b1  = (const float*)d_in[5];
    const float* Wm2 = (const float*)d_in[6];
    const float* b2  = (const float*)d_in[7];
    const float* Wl  = (const float*)d_in[8];
    const float* b3  = (const float*)d_in[9];
    float* out = (float*)d_out;

    // workspace: [0,2MiB) knn indices, [2MiB,10MiB) bf16 x table
    int*    knn = (int*)d_ws;
    bf16_t* xb  = (bf16_t*)((char*)d_ws + (2u << 20));

    cvt_kernel<<<(BATCH * NPTS * DIM) / (256 * 4), 256, 0, stream>>>(x, xb);
    knn_kernel<<<BATCH * (NPTS / 256), 256, 0, stream>>>(pos, knn);
    edgeconv_kernel<<<NBLK, 256, 0, stream>>>(x, xb, Wf, b0, Wm1, b1, Wm2, b2,
                                              Wl, b3, knn, out);
}
